// DiTBlock_57320633532471
// MI455X (gfx1250) — matrix operand order repaired and checked by reading, not compiled, not run on hardware
//
#include <hip/hip_runtime.h>
#include <hip/hip_bf16.h>
#include <math.h>

#define D_  768
#define H_  12
#define HS_ 64
#define B_  8
#define T_  1024
#define TC_ 256
#define FF_ 3072

typedef __bf16 bf16;
typedef __attribute__((ext_vector_type(8)))  bf16  bf16x8;
typedef __attribute__((ext_vector_type(16))) bf16  bf16x16;
typedef __attribute__((ext_vector_type(8)))  float floatx8;
typedef __attribute__((ext_vector_type(4)))  int   i32x4;

#define AS1_ __attribute__((address_space(1)))
#define AS3_ __attribute__((address_space(3)))

// CDNA5 async global->LDS path (ASYNCcnt-tracked), guarded so either toolchain compiles.
#if defined(__AMDGCN__) && __has_builtin(__builtin_amdgcn_global_load_async_to_lds_b128)
#define USE_ASYNC 1
#else
#define USE_ASYNC 0
#endif

__device__ __forceinline__ bf16x8 ld8(const bf16* p) { return *(const bf16x8*)p; }
__device__ __forceinline__ bf16x16 cat8(bf16x8 lo, bf16x8 hi) {
    return __builtin_shufflevector(lo, hi, 0,1,2,3,4,5,6,7,8,9,10,11,12,13,14,15);
}
__device__ __forceinline__ floatx8 wmma_bf16(bf16x16 a, bf16x16 b, floatx8 c) {
    return __builtin_amdgcn_wmma_f32_16x16x32_bf16(false, a, false, b, (short)0, c, false, false);
}

__device__ __forceinline__ void copy16_to_lds(const bf16* g, bf16* l) {
#if USE_ASYNC
    __builtin_amdgcn_global_load_async_to_lds_b128(
        (AS1_ i32x4*)g, (AS3_ i32x4*)l, 0, 0);
#else
    *(bf16x8*)l = ld8(g);
#endif
}

template <int N>
__device__ __forceinline__ void wait_async() {
#if USE_ASYNC
#if __has_builtin(__builtin_amdgcn_s_wait_asynccnt)
    __builtin_amdgcn_s_wait_asynccnt(N);
#else
    asm volatile("s_wait_asynccnt %0" :: "i"(N) : "memory");
#endif
#endif
}

// ---------------------------------------------------------------------------
// Elementwise / packing helpers
// ---------------------------------------------------------------------------
__global__ void copy_f32(const float* __restrict__ in, float* __restrict__ out, size_t n) {
    size_t i = (size_t)blockIdx.x * 256 + threadIdx.x;
    if (i < n) out[i] = in[i];
}

__global__ void f32_to_bf16(const float* __restrict__ in, bf16* __restrict__ out, size_t n) {
    size_t i = (size_t)blockIdx.x * 256 + threadIdx.x;
    if (i < n) out[i] = (bf16)in[i];
}

// W [H, D, HS] fp32 -> Wp [N=D][K=D] bf16  (n = h*HS + e,  Wp[n*D + k] = W[h, k, e])
__global__ void pack_qkv_w(const float* __restrict__ W, bf16* __restrict__ Wp) {
    int i = blockIdx.x * 256 + threadIdx.x;
    if (i >= D_ * D_) return;
    int n = i / D_, kk = i % D_;
    int h = n / HS_, e = n % HS_;
    Wp[i] = (bf16)W[((size_t)h * D_ + kk) * HS_ + e];
}

// W row-major [K][N] fp32 -> Wp [N][K] bf16
__global__ void pack_t_w(const float* __restrict__ W, bf16* __restrict__ Wp, int K, int N) {
    size_t i = (size_t)blockIdx.x * 256 + threadIdx.x;
    if (i >= (size_t)K * N) return;
    int n = (int)(i / K), kk = (int)(i % K);
    Wp[i] = (bf16)W[(size_t)kk * N + n];
}

// ---------------------------------------------------------------------------
// LayerNorm: one block per row of 768, fp32 in -> bf16 out
// ---------------------------------------------------------------------------
__global__ __launch_bounds__(256) void layernorm_to_bf16(
    const float* __restrict__ x, const float* __restrict__ w,
    const float* __restrict__ b, bf16* __restrict__ out)
{
    const int row = blockIdx.x;
    const int tid = threadIdx.x;
    const float* xr = x + (size_t)row * D_;
    float v0 = xr[tid], v1 = xr[tid + 256], v2 = xr[tid + 512];
    float s = v0 + v1 + v2;
    #pragma unroll
    for (int o = 16; o > 0; o >>= 1) s += __shfl_xor(s, o, 32);
    __shared__ float red[8];
    if ((tid & 31) == 0) red[tid >> 5] = s;
    __syncthreads();
    float tot = 0.f;
    #pragma unroll
    for (int i = 0; i < 8; ++i) tot += red[i];
    const float mean = tot * (1.f / 768.f);
    float d0 = v0 - mean, d1 = v1 - mean, d2 = v2 - mean;
    float s2 = d0 * d0 + d1 * d1 + d2 * d2;
    #pragma unroll
    for (int o = 16; o > 0; o >>= 1) s2 += __shfl_xor(s2, o, 32);
    __syncthreads();
    if ((tid & 31) == 0) red[tid >> 5] = s2;
    __syncthreads();
    float tot2 = 0.f;
    #pragma unroll
    for (int i = 0; i < 8; ++i) tot2 += red[i];
    const float rstd = rsqrtf(tot2 * (1.f / 768.f) + 1e-5f);
    out[(size_t)row * D_ + tid      ] = (bf16)(d0 * rstd * w[tid      ] + b[tid      ]);
    out[(size_t)row * D_ + tid + 256] = (bf16)(d1 * rstd * w[tid + 256] + b[tid + 256]);
    out[(size_t)row * D_ + tid + 512] = (bf16)(d2 * rstd * w[tid + 512] + b[tid + 512]);
}

// ---------------------------------------------------------------------------
// Transpose V: v [B*Tk, D] bf16 -> vt [B,H,64,Tk] bf16
// ---------------------------------------------------------------------------
__global__ __launch_bounds__(256) void transpose_v(
    const bf16* __restrict__ v, bf16* __restrict__ vt, int Tk)
{
    __shared__ bf16 tile[64][65];
    const int bh = blockIdx.y;
    const int b = bh / H_, h = bh % H_;
    const int t0 = blockIdx.x * 64;
    for (int i = threadIdx.x; i < 64 * 64; i += 256) {
        int tl = i >> 6, e = i & 63;
        tile[tl][e] = v[((size_t)(b * Tk + t0 + tl)) * D_ + h * HS_ + e];
    }
    __syncthreads();
    for (int i = threadIdx.x; i < 64 * 64; i += 256) {
        int e = i >> 6, tl = i & 63;
        vt[((size_t)(bh * HS_ + e)) * Tk + t0 + tl] = tile[tl][e];
    }
}

// ---------------------------------------------------------------------------
// Tiled bf16 GEMM: C[M,N] = act( (A[M,K] * Bp[N,K]^T + bias) * scale ) (+resid)
// MODE 0: bf16 store   MODE 1: SiLU -> bf16 store   MODE 2: fp32 resid add
// Block tile 128x128 (8 waves, 32x64 per wave), BK=64, double-buffered LDS,
// async global->LDS streaming on CDNA5.
// ---------------------------------------------------------------------------
template <int MODE>
__global__ __launch_bounds__(256) void gemm_bf16(
    const bf16* __restrict__ A, const bf16* __restrict__ Bp,
    const float* __restrict__ bias, float scale,
    const float* __restrict__ resid, void* __restrict__ outv,
    int M, int N, int K)
{
    constexpr int LDT = 72;  // 64 + 8 pad (144B row stride: conflict-free b128 reads)
    __shared__ bf16 As[2][128 * LDT];
    __shared__ bf16 Bs[2][128 * LDT];
    const int tid  = threadIdx.x;
    const int lane = tid & 31, wid = tid >> 5;
    const int bm = blockIdx.y * 128, bn = blockIdx.x * 128;
    const int wm = (wid >> 1) * 32, wn = (wid & 1) * 64;
    const int l16 = lane & 15, hl = lane >> 4;
    const int klo  = hl * 8;    // A-frag K start (per half-wave)
    const int kb16 = hl * 16;   // unused here: B-frag also starts at klo

    floatx8 acc[2][4] = {};

    auto issue_stage = [&](int buf, int k0) {
        #pragma unroll
        for (int i = 0; i < 4; ++i) {
            int c = tid + i * 256;              // 1024 chunks of 8 bf16 per matrix
            int r = c >> 3, cc = (c & 7) * 8;   // 128 rows x 8 chunks (128B/row)
            copy16_to_lds(&A [(size_t)(bm + r) * K + k0 + cc], &As[buf][r * LDT + cc]);
            copy16_to_lds(&Bp[(size_t)(bn + r) * K + k0 + cc], &Bs[buf][r * LDT + cc]);
        }
    };

    const int S = K / 64;
    issue_stage(0, 0);
    for (int s = 0; s < S; ++s) {
        const int buf = s & 1;
        if (s + 1 < S) {
            issue_stage(buf ^ 1, (s + 1) * 64);
            wait_async<8>();      // stage s done; stage s+1 (8 ops) still in flight
        } else {
            wait_async<0>();
        }
        __syncthreads();
        #pragma unroll
        for (int ks = 0; ks < 2; ++ks) {
            const int kof = ks * 32;
            bf16x16 af[2], bfm[4];
            #pragma unroll
            for (int im = 0; im < 2; ++im) {
                const bf16* p = &As[buf][(wm + im * 16 + l16) * LDT + kof + klo];
                af[im] = cat8(ld8(p), ld8(p + 16));
            }
            #pragma unroll
            for (int in = 0; in < 4; ++in) {
                const bf16* p = &Bs[buf][(wn + in * 16 + l16) * LDT + kof + klo];
                bfm[in] = cat8(ld8(p), ld8(p + 16));
            }
            #pragma unroll
            for (int im = 0; im < 2; ++im)
                #pragma unroll
                for (int in = 0; in < 4; ++in)
                    acc[im][in] = wmma_bf16(af[im], bfm[in], acc[im][in]);
        }
        __syncthreads();
    }

    #pragma unroll
    for (int im = 0; im < 2; ++im) {
        #pragma unroll
        for (int in = 0; in < 4; ++in) {
            const int gn  = bn + wn + in * 16 + l16;
            const int gm0 = bm + wm + im * 16 + hl * 8;
            const float bv = bias ? bias[gn] : 0.f;
            #pragma unroll
            for (int r = 0; r < 8; ++r) {
                float v = (acc[im][in][r] + bv) * scale;
                const size_t idx = (size_t)(gm0 + r) * N + gn;
                if (MODE == 0) {
                    ((bf16*)outv)[idx] = (bf16)v;
                } else if (MODE == 1) {
                    float sv = v / (1.f + __expf(-v));     // SiLU
                    ((bf16*)outv)[idx] = (bf16)sv;
                } else {
                    ((float*)outv)[idx] = resid[idx] + v;  // residual add, fp32 out
                }
            }
        }
    }
}

// ---------------------------------------------------------------------------
// Flash attention, transposed orientation: S^T = K*Q^T, O^T = V^T*P^T.
// One wave per 16-query tile; all 8 waves of a block share (b,h), so K and
// V^T blocks are staged once in LDS per 32-key block. In the S^T C-layout a
// lane owns one query (l&15) and 16 keys -> softmax is in-register + one
// xor-16 shuffle; the P B-fragment is built with 8 xor-16 exchanges (no LDS
// bounce). q [B*Tq,D] (pre-scaled), k [B*Tk,D], vt [B,H,64,Tk]; x += O.
// ---------------------------------------------------------------------------
__global__ __launch_bounds__(256) void attn_flash(
    const bf16* __restrict__ q, const bf16* __restrict__ k,
    const bf16* __restrict__ vt, float* __restrict__ x,
    int Tq, int Tk)
{
    constexpr int KLD = 72;               // K tile: 32 keys x 64 hs, padded
    constexpr int VLD = 40;               // V^T tile: 64 hs x 32 keys, padded
    __shared__ bf16 Ks[32 * KLD];
    __shared__ bf16 Vs[64 * VLD];
    const int tid = threadIdx.x;
    const int lane = tid & 31, wid = tid >> 5;
    const int tile = blockIdx.x * 8 + wid;
    const int tilesPerBH = Tq / 16;       // 64 -> all 8 waves share bh
    const int bh = tile / tilesPerBH;
    const int t0 = (tile % tilesPerBH) * 16;
    const int b = bh / H_, h = bh % H_;
    const int l16 = lane & 15, hl = lane >> 4;
    const int klo = hl * 8, kb16 = hl * 16;

    // Q B-fragments for S^T (B[k=hs][n=query] = Q[query][hs]), loaded once
    const bf16* qrow = q + ((size_t)(b * Tq + t0 + l16)) * D_ + h * HS_;
    bf16x16 qf[2];
    qf[0] = cat8(ld8(qrow + klo),      ld8(qrow + klo + 16));
    qf[1] = cat8(ld8(qrow + 32 + klo), ld8(qrow + 32 + klo + 16));

    float rmax = -1e30f, rsum = 0.f;      // per-lane: stats of query l16
    floatx8 o[4] = {};                    // O^T tiles: [16 hs x 16 q] each

    // cooperative staging maps (one 16B chunk per thread per matrix)
    const int kr = tid >> 3, kc = (tid & 7) * 8;   // K: 32 rows x 8 chunks
    const int vr = tid >> 2, vc = (tid & 3) * 8;   // Vt: 64 rows x 4 chunks

    for (int kb = 0; kb < Tk; kb += 32) {
        copy16_to_lds(&k [((size_t)(b * Tk + kb + kr)) * D_ + h * HS_ + kc], &Ks[kr * KLD + kc]);
        copy16_to_lds(&vt[((size_t)(bh * HS_ + vr)) * Tk + kb + vc],         &Vs[vr * VLD + vc]);
        wait_async<0>();
        __syncthreads();

        // S^T tiles: kt=0 -> keys kb..kb+15, kt=1 -> keys kb+16..kb+31
        floatx8 s0 = {}, s1 = {};
        #pragma unroll
        for (int ks = 0; ks < 2; ++ks) {
            const int kof = ks * 32;
            const bf16* p0 = &Ks[l16 * KLD + kof + klo];
            const bf16* p1 = &Ks[(16 + l16) * KLD + kof + klo];
            bf16x16 kf0 = cat8(ld8(p0), ld8(p0 + 16));
            bf16x16 kf1 = cat8(ld8(p1), ld8(p1 + 16));
            s0 = wmma_bf16(kf0, qf[ks], s0);
            s1 = wmma_bf16(kf1, qf[ks], s1);
        }

        // online softmax: lane owns 16 key-scores of query l16
        float m = s0[0];
        #pragma unroll
        for (int r = 1; r < 8; ++r) m = fmaxf(m, s0[r]);
        #pragma unroll
        for (int r = 0; r < 8; ++r) m = fmaxf(m, s1[r]);
        m = fmaxf(m, __shfl_xor(m, 16, 32));          // combine half-waves
        const float nm = fmaxf(rmax, m);
        const float alpha = __expf(rmax - nm);
        rmax = nm;
        float p0v[8], p1v[8];
        float ps = 0.f;
        #pragma unroll
        for (int r = 0; r < 8; ++r) {
            p0v[r] = __expf(s0[r] - nm);
            p1v[r] = __expf(s1[r] - nm);
            ps += p0v[r] + p1v[r];
        }
        ps += __shfl_xor(ps, 16, 32);
        rsum = rsum * alpha + ps;
        #pragma unroll
        for (int t = 0; t < 4; ++t)
            #pragma unroll
            for (int r = 0; r < 8; ++r) o[t][r] *= alpha;

        // Build P B-fragment [32 keys x 16 q]: lane needs keys kb16..kb16+15
        // of its query; the missing 8 live in the xor-16 partner lane.
        float ex[8];
        #pragma unroll
        for (int r = 0; r < 8; ++r)
            ex[r] = __shfl_xor(hl ? p0v[r] : p1v[r], 16, 32);
        bf16x16 pf;
        #pragma unroll
        for (int j = 0; j < 8; ++j) {
            pf[j]     = (bf16)(hl ? ex[j]  : p0v[j]);
            pf[j + 8] = (bf16)(hl ? p1v[j] : ex[j]);
        }

        // O^T += V^T * P^T : A-frags straight from the staged V^T tile
        #pragma unroll
        for (int t = 0; t < 4; ++t) {
            const bf16* vp = &Vs[(t * 16 + l16) * VLD + kb16];
            bf16x16 vf = cat8(ld8(vp), ld8(vp + 8));
            o[t] = wmma_bf16(vf, pf, o[t]);
        }
        __syncthreads();
    }

    // epilogue: normalize; lane writes 8 contiguous floats per hs-tile of its
    // query row (vector-friendly read-modify-write into the fp32 residual)
    const float inv = 1.f / rsum;
    float* xrow = x + ((size_t)(b * Tq + t0 + l16)) * D_ + h * HS_;
    #pragma unroll
    for (int t = 0; t < 4; ++t) {
        float* xp = xrow + t * 16 + hl * 8;
        #pragma unroll
        for (int r = 0; r < 8; ++r) xp[r] += o[t][r] * inv;
    }
}

// ---------------------------------------------------------------------------
// Host-side orchestration
// ---------------------------------------------------------------------------
extern "C" void kernel_launch(void* const* d_in, const int* in_sizes, int n_in,
                              void* d_out, int out_size, void* d_ws, size_t ws_size,
                              hipStream_t stream)
{
    (void)in_sizes; (void)n_in; (void)out_size; (void)ws_size;
    const float* img  = (const float*)d_in[0];
    const float* ctx  = (const float*)d_in[1];
    const float* ln1w = (const float*)d_in[2];
    const float* ln1b = (const float*)d_in[3];
    const float* sWq  = (const float*)d_in[4];
    const float* sbq  = (const float*)d_in[5];
    const float* sWk  = (const float*)d_in[6];
    const float* sbk  = (const float*)d_in[7];
    const float* sWv  = (const float*)d_in[8];
    const float* sbv  = (const float*)d_in[9];
    const float* ln2w = (const float*)d_in[10];
    const float* ln2b = (const float*)d_in[11];
    const float* cWq  = (const float*)d_in[12];
    const float* cbq  = (const float*)d_in[13];
    const float* cWk  = (const float*)d_in[14];
    const float* cbk  = (const float*)d_in[15];
    const float* cWv  = (const float*)d_in[16];
    const float* cbv  = (const float*)d_in[17];
    const float* ln3w = (const float*)d_in[18];
    const float* ln3b = (const float*)d_in[19];
    const float* W1   = (const float*)d_in[20];
    const float* b1   = (const float*)d_in[21];
    const float* W2   = (const float*)d_in[22];
    const float* b2   = (const float*)d_in[23];

    char* ws = (char*)d_ws;
    size_t off = 0;
    auto alloc = [&](size_t bytes) -> char* {
        char* p = ws + off;
        off += (bytes + 255) & ~(size_t)255;
        return p;
    };
    float* xw    = (float*)alloc((size_t)B_ * T_ * D_ * 4);
    bf16* lnout  = (bf16*)alloc((size_t)B_ * T_ * D_ * 2);
    bf16* qb     = (bf16*)alloc((size_t)B_ * T_ * D_ * 2);
    bf16* kbuf   = (bf16*)alloc((size_t)B_ * T_ * D_ * 2);
    bf16* vbuf   = (bf16*)alloc((size_t)B_ * T_ * D_ * 2);
    bf16* vt     = (bf16*)alloc((size_t)B_ * H_ * HS_ * T_ * 2);
    bf16* a1     = qb;   // alias: MLP hidden reuses the (dead) q/k/v/vt region
    bf16* Wqp    = (bf16*)alloc((size_t)D_ * D_ * 2);
    bf16* Wkp    = (bf16*)alloc((size_t)D_ * D_ * 2);
    bf16* Wvp    = (bf16*)alloc((size_t)D_ * D_ * 2);
    bf16* W1p    = (bf16*)alloc((size_t)D_ * FF_ * 2);
    bf16* W2p    = (bf16*)alloc((size_t)D_ * FF_ * 2);
    bf16* ctxb   = (bf16*)alloc((size_t)B_ * TC_ * D_ * 2);

    const size_t NX = (size_t)B_ * T_ * D_;
    const int M  = B_ * T_;    // 8192
    const int Mc = B_ * TC_;   // 2048
    const float qscale = 0.125f;  // HS^-0.5

    // x working copy (residual accumulator)
    copy_f32<<<(unsigned)((NX + 255) / 256), 256, 0, stream>>>(img, xw, NX);

    // ---- self attention ----
    pack_qkv_w<<<(D_ * D_ + 255) / 256, 256, 0, stream>>>(sWq, Wqp);
    pack_qkv_w<<<(D_ * D_ + 255) / 256, 256, 0, stream>>>(sWk, Wkp);
    pack_qkv_w<<<(D_ * D_ + 255) / 256, 256, 0, stream>>>(sWv, Wvp);
    layernorm_to_bf16<<<M, 256, 0, stream>>>(xw, ln1w, ln1b, lnout);
    dim3 gqkv(D_ / 128, M / 128);
    gemm_bf16<0><<<gqkv, 256, 0, stream>>>(lnout, Wqp, sbq, qscale, nullptr, qb,   M, D_, D_);
    gemm_bf16<0><<<gqkv, 256, 0, stream>>>(lnout, Wkp, sbk, 1.0f,   nullptr, kbuf, M, D_, D_);
    gemm_bf16<0><<<gqkv, 256, 0, stream>>>(lnout, Wvp, sbv, 1.0f,   nullptr, vbuf, M, D_, D_);
    transpose_v<<<dim3(T_ / 64, B_ * H_), 256, 0, stream>>>(vbuf, vt, T_);
    attn_flash<<<(B_ * H_ * (T_ / 16)) / 8, 256, 0, stream>>>(qb, kbuf, vt, xw, T_, T_);

    // ---- cross attention ----
    layernorm_to_bf16<<<M, 256, 0, stream>>>(xw, ln2w, ln2b, lnout);
    pack_qkv_w<<<(D_ * D_ + 255) / 256, 256, 0, stream>>>(cWq, Wqp);
    pack_qkv_w<<<(D_ * D_ + 255) / 256, 256, 0, stream>>>(cWk, Wkp);
    pack_qkv_w<<<(D_ * D_ + 255) / 256, 256, 0, stream>>>(cWv, Wvp);
    const size_t NC = (size_t)B_ * TC_ * D_;
    f32_to_bf16<<<(unsigned)((NC + 255) / 256), 256, 0, stream>>>(ctx, ctxb, NC);
    gemm_bf16<0><<<gqkv, 256, 0, stream>>>(lnout, Wqp, cbq, qscale, nullptr, qb, M, D_, D_);
    dim3 gkv(D_ / 128, Mc / 128);
    gemm_bf16<0><<<gkv, 256, 0, stream>>>(ctxb, Wkp, cbk, 1.0f, nullptr, kbuf, Mc, D_, D_);
    gemm_bf16<0><<<gkv, 256, 0, stream>>>(ctxb, Wvp, cbv, 1.0f, nullptr, vbuf, Mc, D_, D_);
    transpose_v<<<dim3(TC_ / 64, B_ * H_), 256, 0, stream>>>(vbuf, vt, TC_);
    attn_flash<<<(B_ * H_ * (T_ / 16)) / 8, 256, 0, stream>>>(qb, kbuf, vt, xw, T_, TC_);

    // ---- MLP ----
    layernorm_to_bf16<<<M, 256, 0, stream>>>(xw, ln3w, ln3b, lnout);
    pack_t_w<<<(unsigned)(((size_t)D_ * FF_ + 255) / 256), 256, 0, stream>>>(W1, W1p, D_, FF_);
    pack_t_w<<<(unsigned)(((size_t)D_ * FF_ + 255) / 256), 256, 0, stream>>>(W2, W2p, FF_, D_);
    dim3 g1(FF_ / 128, M / 128);
    gemm_bf16<1><<<g1, 256, 0, stream>>>(lnout, W1p, b1, 1.0f, nullptr, a1, M, FF_, D_);
    dim3 g2(D_ / 128, M / 128);
    gemm_bf16<2><<<g2, 256, 0, stream>>>(a1, W2p, b2, 1.0f, xw, d_out, M, D_, FF_);
}
